// TopKRouter_51625506897932
// MI455X (gfx1250) — compile-verified, matrix operand order checked by reading, not run on hardware
//
#include <hip/hip_runtime.h>
#include <hip/hip_bf16.h>
#include <math.h>

// -------- CDNA5 WMMA / vector types --------
typedef __attribute__((ext_vector_type(16))) __bf16 v16bf;
typedef __attribute__((ext_vector_type(8)))  float  v8f;
typedef __attribute__((ext_vector_type(4)))  float  f4;   // for nontemporal b128 loads
typedef __attribute__((ext_vector_type(4)))  unsigned int u4;

#define D_DIM  2048
#define E_DIM  64
#define N_TOK  16384
#define TOK_PER_BLOCK 128
#define TOK_PER_WAVE  16
#define KSTEPS 64          // D_DIM / 32
#define PITCH  65          // LDS row pitch (floats) to avoid bank conflicts
#define WP_ELEMS (E_DIM * D_DIM)          // 131072 bf16 = 256 KB

// ---------------------------------------------------------------------------
// Kernel 1: repack W[E=64, D=2048] (fp32, row-major) into bf16 WMMA B-fragments.
// Fragment flat layout: Wp[ ((kt*4 + nt)*32 + lane)*16 + i ]
//   lane: n = nt*16 + (lane&15), half = lane>>4
//   i<8 : k = kt*32 + half*8 + i
//   i>=8: k = kt*32 + 16 + half*8 + (i-8)
// ---------------------------------------------------------------------------
__global__ void pack_W_kernel(const float* __restrict__ W, __bf16* __restrict__ Wp) {
  int gid  = blockIdx.x * blockDim.x + threadIdx.x;   // 0 .. 131071
  int i    = gid & 15;
  int lane = (gid >> 4) & 31;
  int nt   = (gid >> 9) & 3;
  int kt   = gid >> 11;
  int hlf  = lane >> 4;
  int kk   = (i < 8) ? (hlf * 8 + i) : (16 + hlf * 8 + (i - 8));
  int k    = kt * 32 + kk;
  int n    = nt * 16 + (lane & 15);
  Wp[gid]  = (__bf16)W[n * D_DIM + k];
}

// ---------------------------------------------------------------------------
// Kernel 2: router. 256 threads = 8 wave32. W fragments staged once into LDS
// (async global->LDS, ASYNCcnt-tracked), each wave computes a 16-token x
// 64-expert logit tile with v_wmma_f32_16x16x32_bf16, then softmax / top-2 /
// deterministic per-block expert-mass partials.
// Dynamic LDS: 256 KB for W fragments. Static LDS: ~34.5 KB scratch.
// ---------------------------------------------------------------------------
__global__ __launch_bounds__(256, 1)
void router_kernel(const float* __restrict__ x,
                   const float* __restrict__ bvec,
                   const __bf16* __restrict__ Wp,
                   float* __restrict__ out_w,     // [N_TOK*2]
                   float* __restrict__ out_i,     // [N_TOK*2] indices as float
                   float* __restrict__ blockSums) // [gridDim.x * E_DIM]
{
  extern __shared__ __align__(16) char smem_dyn[];          // 256 KB: W fragments
  __bf16* s_W = (__bf16*)smem_dyn;

  __shared__ float s_logits[TOK_PER_BLOCK * PITCH];
  __shared__ float s_max[TOK_PER_BLOCK];
  __shared__ float s_inv[TOK_PER_BLOCK];
  __shared__ float s_b[E_DIM];

  const int tid  = threadIdx.x;
  const int wave = tid >> 5;
  const int lane = tid & 31;
  const int hlf  = lane >> 4;     // which 16-lane half of the wave
  const int mrow = lane & 15;

  if (tid < E_DIM) s_b[tid] = bvec[tid];

  // ---- stage packed W (256 KB) into LDS, once per block ----
  {
    const u4* g4 = (const u4*)Wp;              // 16384 x b128
#if defined(__gfx1250__) && __has_builtin(__builtin_amdgcn_global_load_async_to_lds_b128)
    typedef __attribute__((ext_vector_type(4))) int i4;
#pragma unroll
    for (int j = 0; j < 16384 / 256; ++j) {
      const int idx = j * 256 + tid;
      __builtin_amdgcn_global_load_async_to_lds_b128(
          (__attribute__((address_space(1))) i4*)(g4 + idx),
          (__attribute__((address_space(3))) i4*)((u4*)s_W + idx), 0, 0);
    }
#if __has_builtin(__builtin_amdgcn_s_wait_asynccnt)
    __builtin_amdgcn_s_wait_asynccnt(0);
#else
    asm volatile("s_wait_asynccnt 0x0" ::: "memory");
#endif
#else
    u4* l4 = (u4*)s_W;
    for (int j = 0; j < 16384 / 256; ++j) {
      const int idx = j * 256 + tid;
      l4[idx] = g4[idx];
    }
#endif
  }
  __syncthreads();

  const int token0 = blockIdx.x * TOK_PER_BLOCK + wave * TOK_PER_WAVE;
  const float* __restrict__ xrow = x + (size_t)(token0 + mrow) * D_DIM;
  const v16bf* __restrict__ wpl = (const v16bf*)s_W;

  v8f acc0 = {}, acc1 = {}, acc2 = {}, acc3 = {};

  for (int kt = 0; kt < KSTEPS; ++kt) {
    const int k0 = kt * 32;
    // A fragment: lane half selects K 0-7/16-23 vs 8-15/24-31 (ISA layout).
    // x is streamed exactly once -> non-temporal loads keep it out of L2.
    const f4* plo = (const f4*)(xrow + k0 + hlf * 8);
    const f4* phi = (const f4*)(xrow + k0 + 16 + hlf * 8);
    f4 lo0 = __builtin_nontemporal_load(plo);
    f4 lo1 = __builtin_nontemporal_load(plo + 1);
    f4 hi0 = __builtin_nontemporal_load(phi);
    f4 hi1 = __builtin_nontemporal_load(phi + 1);
    if (kt + 4 < KSTEPS) __builtin_prefetch(xrow + k0 + 128, 0, 0);

    v16bf a;
    a[0]  = (__bf16)lo0[0]; a[1]  = (__bf16)lo0[1];
    a[2]  = (__bf16)lo0[2]; a[3]  = (__bf16)lo0[3];
    a[4]  = (__bf16)lo1[0]; a[5]  = (__bf16)lo1[1];
    a[6]  = (__bf16)lo1[2]; a[7]  = (__bf16)lo1[3];
    a[8]  = (__bf16)hi0[0]; a[9]  = (__bf16)hi0[1];
    a[10] = (__bf16)hi0[2]; a[11] = (__bf16)hi0[3];
    a[12] = (__bf16)hi1[0]; a[13] = (__bf16)hi1[1];
    a[14] = (__bf16)hi1[2]; a[15] = (__bf16)hi1[3];

    const int bbase = (kt * 4) * 32 + lane;      // B fragments from LDS
    v16bf b0 = wpl[bbase +  0];
    v16bf b1 = wpl[bbase + 32];
    v16bf b2 = wpl[bbase + 64];
    v16bf b3 = wpl[bbase + 96];

    acc0 = __builtin_amdgcn_wmma_f32_16x16x32_bf16(false, a, false, b0, (short)0, acc0, false, false);
    acc1 = __builtin_amdgcn_wmma_f32_16x16x32_bf16(false, a, false, b1, (short)0, acc1, false, false);
    acc2 = __builtin_amdgcn_wmma_f32_16x16x32_bf16(false, a, false, b2, (short)0, acc2, false, false);
    acc3 = __builtin_amdgcn_wmma_f32_16x16x32_bf16(false, a, false, b3, (short)0, acc3, false, false);
  }

  // C/D layout: VGPR r holds M=r (lanes 0-15) / M=r+8 (lanes 16-31), N=lane&15
#pragma unroll
  for (int r = 0; r < 8; ++r) {
    const int row = wave * 16 + r + 8 * hlf;     // token within block
    s_logits[row * PITCH +  0 + mrow] = acc0[r];
    s_logits[row * PITCH + 16 + mrow] = acc1[r];
    s_logits[row * PITCH + 32 + mrow] = acc2[r];
    s_logits[row * PITCH + 48 + mrow] = acc3[r];
  }
  __syncthreads();

  // ---- per-token softmax + top-2 (threads 0..127, one token each) ----
  if (tid < TOK_PER_BLOCK) {
    const int token = blockIdx.x * TOK_PER_BLOCK + tid;
    float mx = -3.402823e38f;
#pragma unroll 4
    for (int e = 0; e < E_DIM; ++e) {
      float l = s_logits[tid * PITCH + e] + s_b[e];
      mx = fmaxf(mx, l);
    }
    float sum = 0.f, v0 = -3.402823e38f, v1 = -3.402823e38f;
    int i0 = 0, i1 = 0;
    for (int e = 0; e < E_DIM; ++e) {
      float l = s_logits[tid * PITCH + e] + s_b[e];
      sum += __expf(l - mx);
      if (l > v0)      { v1 = v0; i1 = i0; v0 = l; i0 = e; }
      else if (l > v1) { v1 = l; i1 = e; }
    }
    const float p0 = __expf(v0 - mx);
    const float p1 = __expf(v1 - mx);
    const float inv01 = 1.f / (p0 + p1);
    __builtin_nontemporal_store(p0 * inv01, &out_w[token * 2 + 0]);
    __builtin_nontemporal_store(p1 * inv01, &out_w[token * 2 + 1]);
    __builtin_nontemporal_store((float)i0,  &out_i[token * 2 + 0]);
    __builtin_nontemporal_store((float)i1,  &out_i[token * 2 + 1]);
    s_max[tid] = mx;
    s_inv[tid] = 1.f / sum;
  }
  __syncthreads();

  // ---- deterministic per-block per-expert probability mass ----
  if (tid < E_DIM) {
    float s = 0.f;
    for (int t = 0; t < TOK_PER_BLOCK; ++t)
      s += __expf(s_logits[t * PITCH + tid] + s_b[tid] - s_max[t]) * s_inv[t];
    blockSums[blockIdx.x * E_DIM + tid] = s;
  }
}

// ---------------------------------------------------------------------------
// Kernel 3: aux loss = std(fractions)/mean(fractions) = 64 * std(fractions)
// ---------------------------------------------------------------------------
__global__ void aux_kernel(const float* __restrict__ blockSums,
                           float* __restrict__ out_aux, int nblocks) {
  __shared__ float s[E_DIM];
  const int tid = threadIdx.x;
  float v = 0.f;
  for (int bI = 0; bI < nblocks; ++bI)
    v += blockSums[bI * E_DIM + tid];
  s[tid] = v;
  __syncthreads();
  if (tid == 0) {
    float tot = 0.f;
    for (int e = 0; e < E_DIM; ++e) tot += s[e];
    const float inv = 1.f / tot;
    const float mean = 1.f / 64.f;
    float var = 0.f;
    for (int e = 0; e < E_DIM; ++e) {
      float d = s[e] * inv - mean;
      var += d * d;
    }
    var *= (1.f / 64.f);
    out_aux[0] = sqrtf(var) * 64.f;   // std / mean, mean == 1/64 exactly
  }
}

// ---------------------------------------------------------------------------
extern "C" void kernel_launch(void* const* d_in, const int* in_sizes, int n_in,
                              void* d_out, int out_size, void* d_ws, size_t ws_size,
                              hipStream_t stream) {
  const float* x = (const float*)d_in[0];   // [4,4096,2048] fp32
  const float* W = (const float*)d_in[1];   // [64,2048]     fp32
  const float* b = (const float*)d_in[2];   // [64]          fp32

  float* out = (float*)d_out;               // [32768 w | 32768 idx | 1 aux]

  __bf16* Wp        = (__bf16*)d_ws;                                    // 256 KB
  float*  blockSums = (float*)((char*)d_ws + (size_t)WP_ELEMS * 2);     // 32 KB

  // 1) pack W -> bf16 B-fragments (131072 elems)
  pack_W_kernel<<<512, 256, 0, stream>>>(W, Wp);

  // 2) WMMA router: 128 blocks x 256 threads covers 16384 tokens.
  //    256 KB dynamic LDS holds the packed W fragments (320 KB/WGP on CDNA5).
  const size_t shmem = (size_t)WP_ELEMS * 2;
  router_kernel<<<N_TOK / TOK_PER_BLOCK, 256, shmem, stream>>>(
      x, b, Wp, out, out + N_TOK * 2, blockSums);

  // 3) aux loss reduction (fixed order -> deterministic)
  aux_kernel<<<1, E_DIM, 0, stream>>>(blockSums, out + N_TOK * 4, N_TOK / TOK_PER_BLOCK);
}
